// Attention_6476810683032
// MI455X (gfx1250) — matrix operand order repaired and checked by reading, not compiled, not run on hardware
//
#include <hip/hip_runtime.h>
#include <hip/hip_bf16.h>
#include <math.h>

typedef __bf16 bf16_t;
typedef __attribute__((ext_vector_type(16))) __bf16 v16bf;
typedef __attribute__((ext_vector_type(8)))  float  v8f;

constexpr int cB   = 2;
constexpr int cT   = 2048;
constexpr int cD   = 2048;
constexpr int cNH  = 16;
constexpr int cNKV = 4;
constexpr int cHD  = 128;

// ---------------------------------------------------------------------------
// WMMA helpers (CDNA5 v_wmma_f32_16x16x32_bf16, layouts per ISA 7.12.2)
// ---------------------------------------------------------------------------
__device__ __forceinline__ v8f zero_v8f() {
  v8f z;
#pragma unroll
  for (int i = 0; i < 8; ++i) z[i] = 0.0f;
  return z;
}

__device__ __forceinline__ v8f wmma_bf16(v16bf a, v16bf b, v8f c) {
  return __builtin_amdgcn_wmma_f32_16x16x32_bf16(
      /*neg_a=*/false, a, /*neg_b=*/false, b,
      /*c_mod=*/(short)0, c, /*reuse_a=*/false, /*reuse_b=*/false);
}

// A fragment, 16x32 bf16, source row-major [m][k] with leading dim ld.
// lane L: row m = L%16; VGPR0..3 hold K = 8*(L/16)+0..7, VGPR4..7 hold +16.
__device__ __forceinline__ v16bf load_a_frag(const bf16_t* base, int ld, int lane) {
  const int m = lane & 15, h = lane >> 4;
  const bf16_t* p0 = base + m * ld + h * 8;
  const bf16_t* p1 = p0 + 16;
  v16bf r;
#pragma unroll
  for (int i = 0; i < 8; ++i) { r[i] = p0[i]; r[i + 8] = p1[i]; }
  return r;
}

// B fragment, 32x16 bf16, source stored n-major [n][k] with leading dim ld.
// lane L: col n = L%16; VGPR0..3 hold K = 8*(L/16)+0..7, VGPR4..7 hold +16.
__device__ __forceinline__ v16bf load_b_frag(const bf16_t* base, int ld, int lane) {
  const int n = lane & 15, h = lane >> 4;
  const bf16_t* p0 = base + n * ld + h * 8;
  const bf16_t* p1 = p0 + 16;
  v16bf r;
#pragma unroll
  for (int i = 0; i < 8; ++i) { r[i] = p0[i]; r[i + 8] = p1[i]; }
  return r;
}

// ---------------------------------------------------------------------------
// GEMM: C[f32, MxN] = A[TA, MxK row-major] * B[f32, KxN row-major]
// 128x128 block, BK=32, 8 waves (4 along M x 2 along N).
// LDS double-buffered; next tile's global loads are register-staged and issued
// before the WMMA chain so HBM/L2 latency hides behind the matrix pipe.
// ---------------------------------------------------------------------------
template <typename TA>
__global__ __launch_bounds__(256) void gemm_wmma_kernel(
    const TA* __restrict__ A, const float* __restrict__ B, float* __restrict__ C,
    int M, int N, int K) {
  constexpr int BM = 128, BN = 128, BK = 32;
  __shared__ bf16_t As[2][BM][BK];
  __shared__ bf16_t Bs[2][BN][BK];   // transposed: Bs[buf][n][k]

  const int tid = threadIdx.x, lane = tid & 31, wid = tid >> 5;
  const int wm = wid & 3, wn = wid >> 2;
  const int bm = blockIdx.y * BM, bn = blockIdx.x * BN;

  v8f acc[2][4];
#pragma unroll
  for (int i = 0; i < 2; ++i)
#pragma unroll
    for (int j = 0; j < 4; ++j) acc[i][j] = zero_v8f();

  const int arow = tid >> 1, akb = (tid & 1) * 16;
  const int brow = tid >> 3, bnb = (tid & 7) * 16;

  // prologue: stage k-tile 0 into buffer 0
  {
    const TA* ap = A + (size_t)(bm + arow) * K + akb;
    const float* bp = B + (size_t)brow * N + bn + bnb;
#pragma unroll
    for (int i = 0; i < 16; ++i) As[0][arow][akb + i] = (bf16_t)(float)ap[i];
#pragma unroll
    for (int i = 0; i < 16; ++i) Bs[0][bnb + i][brow] = (bf16_t)bp[i];
  }
  __syncthreads();

  int cur = 0;
  for (int k0 = 0; k0 < K; k0 += BK) {
    const int nxt = cur ^ 1;
    const bool more = (k0 + BK) < K;

    // issue next tile's global loads early (registers)
    float areg[16], breg[16];
    if (more) {
      const TA* ap = A + (size_t)(bm + arow) * K + (k0 + BK) + akb;
      const float* bp = B + (size_t)(k0 + BK + brow) * N + bn + bnb;
#pragma unroll
      for (int i = 0; i < 16; ++i) areg[i] = (float)ap[i];
#pragma unroll
      for (int i = 0; i < 16; ++i) breg[i] = bp[i];
      if (k0 + 2 * BK < K) __builtin_prefetch(ap + BK, 0, 3);  // global_prefetch
    }

    // batch all fragment loads, then the full WMMA chain
    v16bf af[2];
#pragma unroll
    for (int mi = 0; mi < 2; ++mi)
      af[mi] = load_a_frag(&As[cur][wm * 32 + mi * 16][0], BK, lane);
    v16bf bf4[4];
#pragma unroll
    for (int ni = 0; ni < 4; ++ni)
      bf4[ni] = load_b_frag(&Bs[cur][wn * 64 + ni * 16][0], BK, lane);
#pragma unroll
    for (int ni = 0; ni < 4; ++ni)
#pragma unroll
      for (int mi = 0; mi < 2; ++mi)
        acc[mi][ni] = wmma_bf16(af[mi], bf4[ni], acc[mi][ni]);

    // commit next tile to the other LDS buffer
    if (more) {
#pragma unroll
      for (int i = 0; i < 16; ++i) As[nxt][arow][akb + i] = (bf16_t)areg[i];
#pragma unroll
      for (int i = 0; i < 16; ++i) Bs[nxt][bnb + i][brow] = (bf16_t)breg[i];
    }
    __syncthreads();
    cur = nxt;
  }

  const int mlo = (lane >> 4) * 8, nlo = lane & 15;
#pragma unroll
  for (int mi = 0; mi < 2; ++mi)
#pragma unroll
    for (int ni = 0; ni < 4; ++ni) {
      const int m0 = bm + wm * 32 + mi * 16 + mlo;
      const int n0 = bn + wn * 64 + ni * 16 + nlo;
#pragma unroll
      for (int r = 0; r < 8; ++r) C[(size_t)(m0 + r) * N + n0] = acc[mi][ni][r];
    }
}

// ---------------------------------------------------------------------------
// RMSNorm + RoPE + gate sigmoid + bf16 downconvert. One block per (b,t),
// one wave per 128-wide head row. rotate-half partner e <-> e+64 is lane-local.
// ---------------------------------------------------------------------------
__device__ __forceinline__ void rmsnorm_rope_row(
    const float* __restrict__ src, const float* __restrict__ wgt,
    const float* __restrict__ cb, const float* __restrict__ sb,
    int lane, bf16_t* __restrict__ dst) {
  float x[4];
#pragma unroll
  for (int i = 0; i < 4; ++i) x[i] = src[lane + 32 * i];
  float ss = x[0] * x[0] + x[1] * x[1] + x[2] * x[2] + x[3] * x[3];
#pragma unroll
  for (int off = 16; off >= 1; off >>= 1) ss += __shfl_xor(ss, off, 32);
  const float rs = rsqrtf(ss * (1.0f / cHD) + 1e-6f);
  float xn[4];
#pragma unroll
  for (int i = 0; i < 4; ++i) xn[i] = x[i] * rs * wgt[lane + 32 * i];
#pragma unroll
  for (int i = 0; i < 2; ++i) {
    const int e = lane + 32 * i;               // e in [0,64)
    dst[e]      = (bf16_t)(xn[i]     * cb[e]      - xn[i + 2] * sb[e]);
    dst[e + 64] = (bf16_t)(xn[i + 2] * cb[e + 64] + xn[i]     * sb[e + 64]);
  }
}

__global__ __launch_bounds__(256) void qkv_post_kernel(
    const float* __restrict__ qout, const float* __restrict__ kout,
    const float* __restrict__ vout,
    const float* __restrict__ cosb, const float* __restrict__ sinb,
    const float* __restrict__ qw, const float* __restrict__ kw,
    bf16_t* __restrict__ qb, bf16_t* __restrict__ kbuf,
    bf16_t* __restrict__ vbuf, bf16_t* __restrict__ gs) {
  const int bt = blockIdx.x;
  const int lane = threadIdx.x & 31, wid = threadIdx.x >> 5;
  const float* cb = cosb + (size_t)bt * cHD;
  const float* sb = sinb + (size_t)bt * cHD;

  for (int row = wid; row < cNH + 2 * cNKV; row += 8) {
    if (row < cNH) {                           // q head: norm+rope, plus gate
      const int h = row;
      const float* src = qout + (size_t)bt * (cNH * cHD * 2) + (size_t)h * (2 * cHD);
      rmsnorm_rope_row(src, qw, cb, sb, lane, qb + ((size_t)bt * cNH + h) * cHD);
      bf16_t* gdst = gs + ((size_t)bt * cNH + h) * cHD;
#pragma unroll
      for (int i = 0; i < 4; ++i) {
        const int e = lane + 32 * i;
        const float g = src[cHD + e];
        gdst[e] = (bf16_t)(1.0f / (1.0f + __expf(-g)));
      }
    } else if (row < cNH + cNKV) {             // k head: norm+rope
      const int g = row - cNH;
      const float* src = kout + (size_t)bt * (cNKV * cHD) + (size_t)g * cHD;
      rmsnorm_rope_row(src, kw, cb, sb, lane, kbuf + ((size_t)bt * cNKV + g) * cHD);
    } else {                                   // v head: convert only
      const int g = row - cNH - cNKV;
      const float* src = vout + (size_t)bt * (cNKV * cHD) + (size_t)g * cHD;
      bf16_t* dst = vbuf + ((size_t)bt * cNKV + g) * cHD;
#pragma unroll
      for (int i = 0; i < 4; ++i) dst[lane + 32 * i] = (bf16_t)src[lane + 32 * i];
    }
  }
}

// ---------------------------------------------------------------------------
// Flash attention with online softmax. 4 waves, wave w owns q rows [w*16,+16).
// BQ = BKV = 64, causal. Q fragments hoisted out of the KV loop (loop
// invariant); fragment loads batched ahead of each WMMA chain.
// ---------------------------------------------------------------------------
__global__ __launch_bounds__(128) void flash_attn_kernel(
    const bf16_t* __restrict__ Q, const bf16_t* __restrict__ Kc,
    const bf16_t* __restrict__ V, const bf16_t* __restrict__ G,
    bf16_t* __restrict__ O) {
  constexpr int BQ = 64, BKV = 64, QT = cT / BQ;
  const int qt = blockIdx.x % QT;
  const int h  = (blockIdx.x / QT) % cNH;
  const int b  = blockIdx.x / (QT * cNH);
  const int g  = h / (cNH / cNKV);
  const int q0 = qt * BQ;
  const int tid = threadIdx.x, lane = tid & 31, w = tid >> 5;
  const int half = lane >> 4, nlo = lane & 15;

  __shared__ bf16_t Qs[BQ][cHD];
  __shared__ bf16_t Ks[BKV][cHD];
  __shared__ bf16_t Vt[cHD][BKV];       // transposed: Vt[hd][key]
  __shared__ bf16_t Ps[4][16][BKV];     // per-wave P spill (A-frag friendly)

  for (int idx = tid; idx < BQ * cHD; idx += 128) {
    const int r = idx >> 7, c = idx & (cHD - 1);
    Qs[r][c] = Q[(((size_t)b * cT + q0 + r) * cNH + h) * cHD + c];
  }
  __syncthreads();

  // Q fragments are invariant across the KV loop: hold them in registers.
  v16bf qf[4];
#pragma unroll
  for (int kk = 0; kk < 4; ++kk)
    qf[kk] = load_a_frag(&Qs[w * 16][kk * 32], cHD, lane);

  float mrow[8], lrow[8];
#pragma unroll
  for (int r = 0; r < 8; ++r) { mrow[r] = -3.0e38f; lrow[r] = 0.0f; }
  v8f oacc[8];
#pragma unroll
  for (int j = 0; j < 8; ++j) oacc[j] = zero_v8f();

  const float scale = 0.08838834764831845f;  // 1/sqrt(128)

  for (int kb = 0; kb < q0 + BQ; kb += BKV) {
    __syncthreads();
    for (int idx = tid; idx < BKV * cHD; idx += 128) {
      const int r = idx >> 7, c = idx & (cHD - 1);
      const size_t base = (((size_t)b * cT + kb + r) * cNKV + g) * cHD + c;
      Ks[r][c] = Kc[base];
      Vt[c][r] = V[base];
    }
    __syncthreads();

    // S = Q K^T over HD: per 16-key tile, batch 4 B-frags then chain 4 WMMAs
    v8f sacc[4];
#pragma unroll
    for (int j = 0; j < 4; ++j) {
      v16bf bfr[4];
#pragma unroll
      for (int kk = 0; kk < 4; ++kk)
        bfr[kk] = load_b_frag(&Ks[j * 16][kk * 32], cHD, lane);
      v8f s = zero_v8f();
#pragma unroll
      for (int kk = 0; kk < 4; ++kk) s = wmma_bf16(qf[kk], bfr[kk], s);
      sacc[j] = s;
    }

    // causal mask + scale
#pragma unroll
    for (int j = 0; j < 4; ++j)
#pragma unroll
      for (int r = 0; r < 8; ++r) {
        const int rq = q0 + w * 16 + r + 8 * half;
        const int ck = kb + j * 16 + nlo;
        const float sv = sacc[j][r] * scale;
        sacc[j][r] = (ck > rq) ? -3.0e38f : sv;
      }

    // online softmax (row = r + 8*half lives across 16 lanes of one half-wave)
#pragma unroll
    for (int r = 0; r < 8; ++r) {
      float mx = fmaxf(fmaxf(sacc[0][r], sacc[1][r]), fmaxf(sacc[2][r], sacc[3][r]));
#pragma unroll
      for (int off = 8; off >= 1; off >>= 1) mx = fmaxf(mx, __shfl_xor(mx, off, 16));
      const float mnew  = fmaxf(mrow[r], mx);
      const float alpha = __expf(mrow[r] - mnew);
      mrow[r] = mnew;
      float rsum = 0.0f;
#pragma unroll
      for (int j = 0; j < 4; ++j) {
        const float p = __expf(sacc[j][r] - mnew);
        rsum += p;
        Ps[w][r + 8 * half][j * 16 + nlo] = (bf16_t)p;
      }
#pragma unroll
      for (int off = 8; off >= 1; off >>= 1) rsum += __shfl_xor(rsum, off, 16);
      lrow[r] = lrow[r] * alpha + rsum;
#pragma unroll
      for (int j = 0; j < 8; ++j) oacc[j][r] *= alpha;
    }

    // O += P @ V : batch both P fragments, then V fragment pairs per hd tile
    v16bf pf[2];
#pragma unroll
    for (int kc = 0; kc < 2; ++kc)
      pf[kc] = load_a_frag(&Ps[w][0][kc * 32], BKV, lane);
#pragma unroll
    for (int j = 0; j < 8; ++j) {
      v16bf vf0 = load_b_frag(&Vt[j * 16][0],  BKV, lane);
      v16bf vf1 = load_b_frag(&Vt[j * 16][32], BKV, lane);
      oacc[j] = wmma_bf16(pf[0], vf0, oacc[j]);
      oacc[j] = wmma_bf16(pf[1], vf1, oacc[j]);
    }
  }

  // epilogue: 1/l, sigmoid-gate multiply, bf16 store
#pragma unroll
  for (int r = 0; r < 8; ++r) {
    const int rq = q0 + w * 16 + r + 8 * half;
    const float il = 1.0f / lrow[r];
#pragma unroll
    for (int j = 0; j < 8; ++j) {
      const int c = j * 16 + nlo;
      const size_t gi = (((size_t)b * cT + rq) * cNH + h) * cHD + c;
      O[gi] = (bf16_t)(oacc[j][r] * il * (float)G[gi]);
    }
  }
}

// ---------------------------------------------------------------------------
// Launcher
// ---------------------------------------------------------------------------
extern "C" void kernel_launch(void* const* d_in, const int* in_sizes, int n_in,
                              void* d_out, int out_size, void* d_ws, size_t ws_size,
                              hipStream_t stream) {
  (void)in_sizes; (void)n_in; (void)out_size; (void)ws_size;
  const float* hidden = (const float*)d_in[0];
  const float* cosb   = (const float*)d_in[1];
  const float* sinb   = (const float*)d_in[2];
  const float* wq     = (const float*)d_in[3];
  const float* wk     = (const float*)d_in[4];
  const float* wv     = (const float*)d_in[5];
  const float* wo     = (const float*)d_in[6];
  const float* qnw    = (const float*)d_in[7];
  const float* knw    = (const float*)d_in[8];
  float* out = (float*)d_out;

  const int M = cB * cT;                 // 4096 rows
  char* ws = (char*)d_ws;
  size_t off = 0;
  auto alloc = [&](size_t bytes) -> void* {
    void* p = ws + off;
    off += (bytes + 255) & ~(size_t)255;
    return p;
  };
  float*  qout = (float*)alloc((size_t)M * cNH * cHD * 2 * sizeof(float));
  float*  kout = (float*)alloc((size_t)M * cNKV * cHD * sizeof(float));
  float*  vout = (float*)alloc((size_t)M * cNKV * cHD * sizeof(float));
  bf16_t* qb   = (bf16_t*)alloc((size_t)M * cNH * cHD * sizeof(bf16_t));
  bf16_t* kbuf = (bf16_t*)alloc((size_t)M * cNKV * cHD * sizeof(bf16_t));
  bf16_t* vbuf = (bf16_t*)alloc((size_t)M * cNKV * cHD * sizeof(bf16_t));
  bf16_t* gs   = (bf16_t*)alloc((size_t)M * cNH * cHD * sizeof(bf16_t));
  bf16_t* ao   = (bf16_t*)alloc((size_t)M * cNH * cHD * sizeof(bf16_t));

  dim3 blk(256);
  // QKV projections (A = fp32 hidden, converted to bf16 during LDS staging)
  gemm_wmma_kernel<float><<<dim3((cNH * cHD * 2) / 128, M / 128), blk, 0, stream>>>(
      hidden, wq, qout, M, cNH * cHD * 2, cD);
  gemm_wmma_kernel<float><<<dim3((cNKV * cHD) / 128, M / 128), blk, 0, stream>>>(
      hidden, wk, kout, M, cNKV * cHD, cD);
  gemm_wmma_kernel<float><<<dim3((cNKV * cHD) / 128, M / 128), blk, 0, stream>>>(
      hidden, wv, vout, M, cNKV * cHD, cD);

  // RMSNorm + RoPE + gate
  qkv_post_kernel<<<M, 256, 0, stream>>>(qout, kout, vout, cosb, sinb, qnw, knw,
                                         qb, kbuf, vbuf, gs);

  // Flash attention (causal, GQA, gated epilogue)
  flash_attn_kernel<<<cB * cNH * (cT / 64), 128, 0, stream>>>(qb, kbuf, vbuf, gs, ao);

  // Output projection (A = bf16 gated attention output)
  gemm_wmma_kernel<bf16_t><<<dim3(cD / 128, M / 128), blk, 0, stream>>>(
      ao, wo, out, M, cD, cNH * cHD);
}
